// ACmix_31928786878541
// MI455X (gfx1250) — compile-verified, matrix operand order checked by reading, not run on hardware
//
#include <hip/hip_runtime.h>

// ---------------------------------------------------------------------------
// CDNA5 (gfx1250) ACmix: bf16 WMMA GEMMs + WMMA window attention.
// wave32; 256-thread blocks = 8 waves.
// ---------------------------------------------------------------------------

typedef __attribute__((ext_vector_type(16))) __bf16 v16bf;
typedef __attribute__((ext_vector_type(8)))  float  v8f;

union FragBF {
    v16bf v;
    uint4 q[2];
    __bf16 h[16];
};

__device__ __forceinline__ v8f wmma_bf16(v16bf a, v16bf b, v8f c) {
    // D = A(16x32 bf16) * B(32x16 bf16) + C(16x16 f32)
    return __builtin_amdgcn_wmma_f32_16x16x32_bf16(
        /*neg_a=*/false, a, /*neg_b=*/false, b,
        /*c_mod=*/(short)0, c, /*reuse_a=*/false, /*reuse_b=*/false);
}

__device__ __forceinline__ void cvt8(__bf16* dst, const float* src, float scale) {
    float4 a = *(const float4*)(src);
    float4 c = *(const float4*)(src + 4);
    dst[0] = (__bf16)(a.x * scale);
    dst[1] = (__bf16)(a.y * scale);
    dst[2] = (__bf16)(a.z * scale);
    dst[3] = (__bf16)(a.w * scale);
    dst[4] = (__bf16)(c.x * scale);
    dst[5] = (__bf16)(c.y * scale);
    dst[6] = (__bf16)(c.z * scale);
    dst[7] = (__bf16)(c.w * scale);
}

// Problem constants
#define B_SZ   16
#define HWSZ   3136                // 56*56
#define NHEAD  8
#define NWIN   49
#define MROWS  (B_SZ * HWSZ)       // 50176 == 3136 * 16  (exact 16-tile fit)
#define MT_ALL (MROWS / 16)        // 3136
#define KT_QKV 8                   // K=256 -> 8 k-tiles of 32
#define NT_QKV 48                  // N=768 -> 48 n-tiles of 16
#define NT_PRJ 16                  // N=256

// Workspace layout (bytes)
#define SZ_XPACK  ((size_t)MT_ALL * KT_QKV * 1024)          // 25,690,112
#define SZ_WQKV   ((size_t)KT_QKV * NT_QKV * 1024)          //    393,216
#define SZ_WPROJ  ((size_t)KT_QKV * NT_PRJ * 1024)          //    131,072
#define SZ_QKV    ((size_t)MROWS * 768 * 4)                 // 154,140,672
#define SZ_FCONV  ((size_t)B_SZ * 288 * HWSZ * 4)           //  57,802,752
#define SZ_CONV   ((size_t)B_SZ * 256 * HWSZ * 4)           //  51,380,224
#define SZ_OPACK  SZ_XPACK
#define OFF_XPACK ((size_t)0)
#define OFF_WQKV  (OFF_XPACK + SZ_XPACK)
#define OFF_WPROJ (OFF_WQKV + SZ_WQKV)
#define OFF_QKV   (OFF_WPROJ + SZ_WPROJ)
#define OFF_FCONV (OFF_QKV + SZ_QKV)
#define OFF_CONV  (OFF_FCONV + SZ_FCONV)
#define OFF_OPACK (OFF_CONV + SZ_CONV)
#define OFF_PROJO (OFF_OPACK + SZ_OPACK)

__device__ __forceinline__ int win_row(int b, int wh, int ww, int n) {
    int r = n / 7, c = n % 7;
    return (b * 56 + wh * 7 + r) * 56 + (ww * 7 + c);
}

// ---------------------------------------------------------------------------
// Pack x (B,C,H,W) f32 -> bf16 A-fragment layout:
//   chunk (mt, kt): 32 lanes x 16 bf16; lane holds row m = mt*16 + lane%16,
//   elems [0..7]  = K = kt*32 + (lane<16?0:8) + 0..7,
//   elems [8..15] = K = kt*32 + 16 + (lane<16?0:8) + 0..7.
// ---------------------------------------------------------------------------
__global__ __launch_bounds__(256) void pack_x_kernel(const float* __restrict__ x,
                                                     __bf16* __restrict__ out) {
    int tid  = blockIdx.x * 256 + threadIdx.x;   // (mt, kt, lane)
    int lane = tid & 31;
    int kt   = (tid >> 5) & 7;
    int mt   = tid >> 8;
    int lh   = lane & 15, hi = lane >> 4;
    int m    = mt * 16 + lh;
    int b    = m / HWSZ, hw = m % HWSZ;
    int k0   = kt * 32 + hi * 8;
    const float* src = x + (size_t)b * 256 * HWSZ + hw;
    __bf16* dst = out + (size_t)tid * 16;
#pragma unroll
    for (int j = 0; j < 8; ++j) dst[j]     = (__bf16)src[(size_t)(k0 + j) * HWSZ];
#pragma unroll
    for (int j = 0; j < 8; ++j) dst[8 + j] = (__bf16)src[(size_t)(k0 + 16 + j) * HWSZ];
}

// ---------------------------------------------------------------------------
// Pack weights into bf16 B-fragment layout: chunk (kt, nt): lane holds col
// n = nt*16 + lane%16; elem e -> K = kt*32 + (e<8?e:e+8) + (lane<16?0:8).
// transpose==0: W is [K][N]; transpose==1: W is [N][K] (for proj_w: y = x W^T).
// ---------------------------------------------------------------------------
__global__ __launch_bounds__(256) void pack_w_kernel(const float* __restrict__ W,
                                                     __bf16* __restrict__ out,
                                                     int K, int N, int Ntiles,
                                                     int transpose) {
    int tid  = blockIdx.x * 256 + threadIdx.x;
    int lane = tid & 31;
    int cnt  = tid >> 5;                // kt*Ntiles + nt
    int nt   = cnt % Ntiles;
    int kt   = cnt / Ntiles;
    int lh   = lane & 15, hi = lane >> 4;
    int n    = nt * 16 + lh;
    __bf16* dst = out + (size_t)tid * 16;
#pragma unroll
    for (int e = 0; e < 16; ++e) {
        int kk = kt * 32 + (e < 8 ? e : e + 8) + hi * 8;
        float v = transpose ? W[(size_t)n * K + kk] : W[(size_t)kk * N + n];
        dst[e] = (__bf16)v;
    }
}

// ---------------------------------------------------------------------------
// GEMM: C[M][ldc] = A(bf16, packed) * B(bf16, packed) + bias[N]
// Block = 8 waves: 4(M) x 2(N); wave tile = 32x32 (2x2 WMMA accumulators).
// Software double-buffered fragments: k+1's loads issue before k's WMMAs.
// grid.x = Ntiles/4, grid.y = Mtiles/8.   KT is compile-time for full unroll.
// ---------------------------------------------------------------------------
template <int KT>
__global__ __launch_bounds__(256) void gemm_bf16_kernel(
        const __bf16* __restrict__ Apack, const __bf16* __restrict__ Bpack,
        const float* __restrict__ bias, float* __restrict__ C,
        int Ntiles, int ldc) {
    const int lane = threadIdx.x & 31;
    const int wave = threadIdx.x >> 5;
    const int lh = lane & 15, hi = lane >> 4;
    const int mw = wave & 3, nw = wave >> 2;
    const int nt0 = blockIdx.x * 4 + nw * 2;     // 2 n-subtiles
    const int mt0 = blockIdx.y * 8 + mw * 2;     // 2 m-subtiles

    const float bv0 = bias[(nt0 + 0) * 16 + lh];
    const float bv1 = bias[(nt0 + 1) * 16 + lh];
    v8f acc00, acc01, acc10, acc11;
#pragma unroll
    for (int i = 0; i < 8; ++i) {
        acc00[i] = bv0; acc01[i] = bv1;
        acc10[i] = bv0; acc11[i] = bv1;
    }

    const __bf16* aB0 = Apack + (size_t)(mt0 + 0) * KT * 512 + lane * 16;
    const __bf16* aB1 = Apack + (size_t)(mt0 + 1) * KT * 512 + lane * 16;
    const __bf16* bB0 = Bpack + (size_t)(nt0 + 0) * 512 + lane * 16;
    const __bf16* bB1 = Bpack + (size_t)(nt0 + 1) * 512 + lane * 16;
    const size_t bStride = (size_t)Ntiles * 512;

    FragBF a0, a1, b0, b1;
    a0.q[0] = *(const uint4*)(aB0);     a0.q[1] = *(const uint4*)(aB0 + 8);
    a1.q[0] = *(const uint4*)(aB1);     a1.q[1] = *(const uint4*)(aB1 + 8);
    b0.q[0] = *(const uint4*)(bB0);     b0.q[1] = *(const uint4*)(bB0 + 8);
    b1.q[0] = *(const uint4*)(bB1);     b1.q[1] = *(const uint4*)(bB1 + 8);

#pragma unroll
    for (int kt = 0; kt < KT; ++kt) {
        FragBF na0, na1, nb0, nb1;
        if (kt + 1 < KT) {
            const __bf16* pa0 = aB0 + (size_t)(kt + 1) * 512;
            const __bf16* pa1 = aB1 + (size_t)(kt + 1) * 512;
            const __bf16* pb0 = bB0 + (size_t)(kt + 1) * bStride;
            const __bf16* pb1 = bB1 + (size_t)(kt + 1) * bStride;
            na0.q[0] = *(const uint4*)(pa0); na0.q[1] = *(const uint4*)(pa0 + 8);
            na1.q[0] = *(const uint4*)(pa1); na1.q[1] = *(const uint4*)(pa1 + 8);
            nb0.q[0] = *(const uint4*)(pb0); nb0.q[1] = *(const uint4*)(pb0 + 8);
            nb1.q[0] = *(const uint4*)(pb1); nb1.q[1] = *(const uint4*)(pb1 + 8);
            if (kt + 2 < KT) {
                __builtin_prefetch(pa0 + 512, 0, 3);   // near-cache prefetch
                __builtin_prefetch(pa1 + 512, 0, 3);
            }
        }
        acc00 = wmma_bf16(a0.v, b0.v, acc00);
        acc01 = wmma_bf16(a0.v, b1.v, acc01);
        acc10 = wmma_bf16(a1.v, b0.v, acc10);
        acc11 = wmma_bf16(a1.v, b1.v, acc11);
        if (kt + 1 < KT) { a0 = na0; a1 = na1; b0 = nb0; b1 = nb1; }
    }

    const int col0 = (nt0 + 0) * 16 + lh;
    const int col1 = (nt0 + 1) * 16 + lh;
#pragma unroll
    for (int r = 0; r < 8; ++r) {
        int row0 = mt0 * 16 + r + hi * 8;
        int row1 = row0 + 16;
        C[(size_t)row0 * ldc + col0] = acc00[r];
        C[(size_t)row0 * ldc + col1] = acc01[r];
        C[(size_t)row1 * ldc + col0] = acc10[r];
        C[(size_t)row1 * ldc + col1] = acc11[r];
    }
}

// ---------------------------------------------------------------------------
// Conv branch stage 1: fc_out mix.  thread = (b, hw, d), 9 outputs from 24.
// fconv layout: [b][d*9+o][hw]
// ---------------------------------------------------------------------------
__global__ __launch_bounds__(256) void fc_kernel(const float* __restrict__ qkv,
                                                 const float* __restrict__ fc_w,
                                                 const float* __restrict__ fc_b,
                                                 float* __restrict__ fconv) {
    int tid = blockIdx.x * 256 + threadIdx.x;
    int d = tid & 31;
    int t = tid >> 5;                   // b*HWSZ + hw  == global row m
    int b = t / HWSZ, hw = t % HWSZ;
    float acc[9];
#pragma unroll
    for (int o = 0; o < 9; ++o) acc[o] = fc_b[o];
    const float* row = qkv + (size_t)t * 768 + d;
#pragma unroll
    for (int g = 0; g < 24; ++g) {
        float v = row[g * 32];
#pragma unroll
        for (int o = 0; o < 9; ++o) acc[o] = fmaf(fc_w[o * 24 + g], v, acc[o]);
    }
#pragma unroll
    for (int o = 0; o < 9; ++o)
        fconv[((size_t)b * 288 + d * 9 + o) * HWSZ + hw] = acc[o];
}

// ---------------------------------------------------------------------------
// Conv branch stage 2: grouped 3x3 conv (groups = 32, 9 in / 8 out per grp)
// ---------------------------------------------------------------------------
__global__ __launch_bounds__(256) void dwconv_kernel(const float* __restrict__ fconv,
                                                     const float* __restrict__ dep_w,
                                                     float* __restrict__ convout) {
    int tid = blockIdx.x * 256 + threadIdx.x;   // ((b*256+oc)*HWSZ + hw)
    int hw = tid % HWSZ;
    int t  = tid / HWSZ;
    int oc = t % 256, b = t / 256;
    int h = hw / 56, w = hw % 56;
    int gi = oc >> 3;
    const float* base  = fconv + ((size_t)b * 288 + gi * 9) * HWSZ;
    const float* wbase = dep_w + (size_t)oc * 81;
    float acc = 0.f;
#pragma unroll
    for (int o = 0; o < 9; ++o)
#pragma unroll
        for (int kh = 0; kh < 3; ++kh) {
            int hh = h + kh - 1;
            if (hh < 0 || hh >= 56) continue;
#pragma unroll
            for (int kw = 0; kw < 3; ++kw) {
                int wp = w + kw - 1;
                if (wp < 0 || wp >= 56) continue;
                acc = fmaf(wbase[o * 9 + kh * 3 + kw],
                           base[(size_t)o * HWSZ + hh * 56 + wp], acc);
            }
        }
    convout[tid] = acc;
}

// ---------------------------------------------------------------------------
// Window attention: block = window b_, wave = head.
// S = Q*K^T via 4 WMMA (rpb bias + -inf mask preloaded in C-accumulator),
// raw S transposed C->A layout through LDS (f32), softmax fully in-register
// per lane (one row per lane; single xor-16 shuffle for max and sum),
// O = P*V via 4 WMMA, scattered directly into packed-A bf16 layout for the
// proj GEMM.
// ---------------------------------------------------------------------------
__global__ __launch_bounds__(256) void attn_kernel(const float* __restrict__ qkv,
                                                   const float* __restrict__ rpb,
                                                   __bf16* __restrict__ Opack) {
    __shared__ __align__(16) float sS[NHEAD][16][64];   // 32 KB: raw S transpose
    __shared__ float sRpb[169 * NHEAD];                 // 5.4 KB: bias table
    const int b_   = blockIdx.x;
    const int head = threadIdx.x >> 5;
    const int lane = threadIdx.x & 31;
    const int lh = lane & 15, hi = lane >> 4;
    const int b = b_ >> 6, wt = b_ & 63, wh = wt >> 3, ww = wt & 7;

    for (int i = threadIdx.x; i < 169 * NHEAD; i += 256) sRpb[i] = rpb[i];
    __syncthreads();

    // K fragments: B-layout, key = nt*16+lh per lane, d packed per element
    FragBF kf[4];
    int rk[4], ck[4], kvalid[4];
#pragma unroll
    for (int nt = 0; nt < 4; ++nt) {
        int key = nt * 16 + lh;
        kvalid[nt] = key < NWIN;
        rk[nt] = key / 7;
        ck[nt] = key % 7;
        if (kvalid[nt]) {
            const float* src = qkv + (size_t)win_row(b, wh, ww, key) * 768
                             + 256 + head * 32 + hi * 8;
            cvt8(kf[nt].h, src, 1.0f);
            cvt8(kf[nt].h + 8, src + 16, 1.0f);
        } else {
#pragma unroll
            for (int e = 0; e < 16; ++e) kf[nt].h[e] = (__bf16)0.f;
        }
    }

    // V fragments: B-layout, n = d, K = key (padded >=49 -> 0)
    FragBF vf[2][2];
#pragma unroll
    for (int nt2 = 0; nt2 < 2; ++nt2)
#pragma unroll
        for (int ks = 0; ks < 2; ++ks) {
            int d = nt2 * 16 + lh;
#pragma unroll
            for (int e = 0; e < 16; ++e) {
                int key = ks * 32 + (e < 8 ? e : e + 8) + hi * 8;
                float v = 0.f;
                if (key < NWIN)
                    v = qkv[(size_t)win_row(b, wh, ww, key) * 768 + 512 + head * 32 + d];
                vf[nt2][ks].h[e] = (__bf16)v;
            }
        }

    const float scale = 0.1767766953f;   // 1/sqrt(HD)
    for (int mt = 0; mt < 4; ++mt) {
        // Q fragment (A-layout, scale folded in)
        FragBF qf;
        int qrow = mt * 16 + lh;
        if (qrow < NWIN) {
            const float* src = qkv + (size_t)win_row(b, wh, ww, qrow) * 768
                             + head * 32 + hi * 8;
            cvt8(qf.h, src, scale);
            cvt8(qf.h + 8, src + 16, scale);
        } else {
#pragma unroll
            for (int e = 0; e < 16; ++e) qf.h[e] = (__bf16)0.f;
        }

        // query coords for the 8 C-layout rows this lane touches
        int rq[8], cq[8], qvalid[8];
#pragma unroll
        for (int r = 0; r < 8; ++r) {
            int qq = mt * 16 + r + hi * 8;
            qvalid[r] = qq < NWIN;
            rq[r] = qq / 7;
            cq[r] = qq % 7;
        }

        // S accumulators seeded with relative-position bias / -inf mask
        v8f s[4];
#pragma unroll
        for (int nt = 0; nt < 4; ++nt)
#pragma unroll
            for (int r = 0; r < 8; ++r) {
                float bini = -1e30f;
                if (qvalid[r] && kvalid[nt])
                    bini = sRpb[((rq[r] - rk[nt] + 6) * 13 + (cq[r] - ck[nt] + 6))
                                * NHEAD + head];
                s[nt][r] = bini;
            }
#pragma unroll
        for (int nt = 0; nt < 4; ++nt) s[nt] = wmma_bf16(qf.v, kf[nt].v, s[nt]);

        // Transpose raw S (C-layout -> A-layout rows) through wave-private LDS
#pragma unroll
        for (int nt = 0; nt < 4; ++nt)
#pragma unroll
            for (int r = 0; r < 8; ++r)
                sS[head][r + hi * 8][nt * 16 + lh] = s[nt][r];

        // Each lane now owns row lh (its half of the 64 keys, in frag order)
        union { float4 f4[8]; float v[32]; } u;
        const float* base = &sS[head][lh][0];
        u.f4[0] = *(const float4*)(base + hi * 8);
        u.f4[1] = *(const float4*)(base + hi * 8 + 4);
        u.f4[2] = *(const float4*)(base + 16 + hi * 8);
        u.f4[3] = *(const float4*)(base + 16 + hi * 8 + 4);
        u.f4[4] = *(const float4*)(base + 32 + hi * 8);
        u.f4[5] = *(const float4*)(base + 32 + hi * 8 + 4);
        u.f4[6] = *(const float4*)(base + 48 + hi * 8);
        u.f4[7] = *(const float4*)(base + 48 + hi * 8 + 4);

        // In-register softmax: single cross-half shuffle for max and for sum
        float mx = u.v[0];
#pragma unroll
        for (int i = 1; i < 32; ++i) mx = fmaxf(mx, u.v[i]);
        mx = fmaxf(mx, __shfl_xor(mx, 16, 32));
        float sum = 0.f;
#pragma unroll
        for (int i = 0; i < 32; ++i) { u.v[i] = __expf(u.v[i] - mx); sum += u.v[i]; }
        sum += __shfl_xor(sum, 16, 32);
        const float inv = 1.0f / sum;

        // P fragments (A-layout) straight from registers
        FragBF pf[2];
#pragma unroll
        for (int e = 0; e < 16; ++e) {
            pf[0].h[e] = (__bf16)(u.v[e] * inv);
            pf[1].h[e] = (__bf16)(u.v[16 + e] * inv);
        }

        // O = P * V
        v8f o0, o1;
#pragma unroll
        for (int i = 0; i < 8; ++i) { o0[i] = 0.f; o1[i] = 0.f; }
        o0 = wmma_bf16(pf[0].v, vf[0][0].v, o0);
        o0 = wmma_bf16(pf[1].v, vf[0][1].v, o0);
        o1 = wmma_bf16(pf[0].v, vf[1][0].v, o1);
        o1 = wmma_bf16(pf[1].v, vf[1][1].v, o1);

        // Scatter O straight into packed-A bf16 layout for the proj GEMM
#pragma unroll
        for (int nt2 = 0; nt2 < 2; ++nt2) {
            const v8f& oa = nt2 ? o1 : o0;
#pragma unroll
            for (int r = 0; r < 8; ++r) {
                if (qvalid[r]) {
                    int qq = mt * 16 + r + hi * 8;
                    int mrow = b_ * NWIN + qq;
                    int c = head * 32 + nt2 * 16 + lh;
                    int koff = c & 31;
                    int lp = (mrow & 15) + (((koff & 15) < 8) ? 0 : 16);
                    int e  = (koff & 7) + ((koff >= 16) ? 8 : 0);
                    size_t chunk = ((size_t)(mrow >> 4) * KT_QKV + head) * 32 + lp;
                    Opack[chunk * 16 + e] = (__bf16)oa[r];
                }
            }
        }
    }
}

// ---------------------------------------------------------------------------
// Combine: out[b][c][h][w] = rate1 * proj(window token) + rate2 * conv
// ---------------------------------------------------------------------------
__global__ __launch_bounds__(256) void combine_kernel(const float* __restrict__ projo,
                                                      const float* __restrict__ convout,
                                                      const float* __restrict__ rate1,
                                                      const float* __restrict__ rate2,
                                                      float* __restrict__ out) {
    int tid = blockIdx.x * 256 + threadIdx.x;   // ((b*256+c)*HWSZ + hw)
    int hw = tid % HWSZ;
    int t  = tid / HWSZ;
    int c = t % 256, b = t / 256;
    int h = hw / 56, w = hw % 56;
    int wh = h / 7, r = h % 7, ww = w / 7, cc = w % 7;
    int b_ = b * 64 + wh * 8 + ww;
    int n  = r * 7 + cc;
    size_t mrow = (size_t)b_ * NWIN + n;
    out[tid] = rate1[0] * projo[mrow * 256 + c] + rate2[0] * convout[tid];
}

// ---------------------------------------------------------------------------
extern "C" void kernel_launch(void* const* d_in, const int* in_sizes, int n_in,
                              void* d_out, int out_size, void* d_ws, size_t ws_size,
                              hipStream_t stream) {
    const float* x      = (const float*)d_in[0];
    const float* qkv_w  = (const float*)d_in[1];
    const float* qkv_b  = (const float*)d_in[2];
    const float* fc_w   = (const float*)d_in[3];
    const float* fc_b   = (const float*)d_in[4];
    const float* dep_w  = (const float*)d_in[5];
    const float* proj_w = (const float*)d_in[6];
    const float* proj_b = (const float*)d_in[7];
    const float* rpb    = (const float*)d_in[8];
    const float* rate1  = (const float*)d_in[9];
    const float* rate2  = (const float*)d_in[10];
    (void)in_sizes; (void)n_in; (void)out_size; (void)ws_size;

    char* ws = (char*)d_ws;
    __bf16* xpack   = (__bf16*)(ws + OFF_XPACK);
    __bf16* wqkv    = (__bf16*)(ws + OFF_WQKV);
    __bf16* wproj   = (__bf16*)(ws + OFF_WPROJ);
    float*  qkvbuf  = (float*)(ws + OFF_QKV);
    float*  fconv   = (float*)(ws + OFF_FCONV);
    float*  convout = (float*)(ws + OFF_CONV);
    __bf16* opack   = (__bf16*)(ws + OFF_OPACK);
    float*  projo   = (float*)(ws + OFF_PROJO);
    float*  out     = (float*)d_out;

    // 1. pack operands to bf16 fragment layouts
    pack_x_kernel<<<MT_ALL * KT_QKV * 32 / 256, 256, 0, stream>>>(x, xpack);
    pack_w_kernel<<<KT_QKV * NT_QKV * 32 / 256, 256, 0, stream>>>(qkv_w, wqkv, 256, 768, NT_QKV, 0);
    pack_w_kernel<<<KT_QKV * NT_PRJ * 32 / 256, 256, 0, stream>>>(proj_w, wproj, 256, 256, NT_PRJ, 1);

    // 2. QKV GEMM: (50176x256) x (256x768) + b
    gemm_bf16_kernel<KT_QKV><<<dim3(NT_QKV / 4, MT_ALL / 8), 256, 0, stream>>>(
        xpack, wqkv, qkv_b, qkvbuf, NT_QKV, 768);

    // 3. conv branch
    fc_kernel<<<(B_SZ * HWSZ * 32) / 256, 256, 0, stream>>>(qkvbuf, fc_w, fc_b, fconv);
    dwconv_kernel<<<(B_SZ * 256 * HWSZ) / 256, 256, 0, stream>>>(fconv, dep_w, convout);

    // 4. window attention (writes packed bf16 O for proj GEMM)
    attn_kernel<<<B_SZ * 64, 256, 0, stream>>>(qkvbuf, rpb, opack);

    // 5. proj GEMM: (50176x256) x (256x256)^T + b
    gemm_bf16_kernel<KT_QKV><<<dim3(NT_PRJ / 4, MT_ALL / 8), 256, 0, stream>>>(
        opack, wproj, proj_b, projo, NT_PRJ, 256);

    // 6. combine branches -> NCHW
    combine_kernel<<<(B_SZ * 256 * HWSZ) / 256, 256, 0, stream>>>(
        projo, convout, rate1, rate2, out);
}
